// HyperbolicGNN_13125420056910
// MI455X (gfx1250) — matrix operand order repaired and checked by reading, not compiled, not run on hardware
//
#include <hip/hip_runtime.h>
#include <hip/hip_bf16.h>
#include <math.h>

// ---------------------------------------------------------------------------
// Hyperbolic GNN (Poincare ball, c=1) for MI455X / gfx1250.
//   layer: t=logmap0(x); h = t@W^T + b; y=expmap0(h); acc = segment_sum(y[src],dst)
// Tangent buffer kept in bf16; GEMM via v_wmma_f32_16x16x32_bf16 (wave32).
// ---------------------------------------------------------------------------

typedef __attribute__((ext_vector_type(16))) __bf16        v16bf;
typedef __attribute__((ext_vector_type(4)))  __bf16        v4bf;
typedef __attribute__((ext_vector_type(8)))  float         v8f;
typedef __attribute__((ext_vector_type(4)))  unsigned int  v4u;
typedef __attribute__((ext_vector_type(8)))  unsigned int  v8u;

#define DIMN   128
#define EPSF   1e-15f
#define MAXNF  (1.0f - 1e-5f)

__device__ __forceinline__ float red16(float x) {   // reduce within 16-lane half
  x += __shfl_xor(x, 1, 32);
  x += __shfl_xor(x, 2, 32);
  x += __shfl_xor(x, 4, 32);
  x += __shfl_xor(x, 8, 32);
  return x;
}
__device__ __forceinline__ float red32(float x) {   // full-wave reduce
  x = red16(x);
  x += __shfl_xor(x, 16, 32);
  return x;
}
__device__ __forceinline__ float artanhf_(float z) {
  return 0.5f * logf((1.0f + z) / (1.0f - z));
}

// ---------------- f32 -> bf16 weight conversion ----------------------------
__global__ void cvt_bf16_kernel(const float* __restrict__ src,
                                __bf16* __restrict__ dst, int n) {
  int i = blockIdx.x * blockDim.x + threadIdx.x;
  if (i < n) dst[i] = (__bf16)src[i];
}

// ---------------- logmap0 (+ optional ReLU) -> bf16 tangent ----------------
// one wave per node, 4 floats per lane (128 dims)
__global__ void tangent_kernel(const float* __restrict__ x,
                               v4bf* __restrict__ t,
                               int n_nodes, int do_relu) {
  int wave = (blockIdx.x * blockDim.x + threadIdx.x) >> 5;
  int lane = threadIdx.x & 31;
  if (wave >= n_nodes) return;
  float4 v = ((const float4*)(x + (size_t)wave * DIMN))[lane];
  if (do_relu) {
    v.x = fmaxf(v.x, 0.0f); v.y = fmaxf(v.y, 0.0f);
    v.z = fmaxf(v.z, 0.0f); v.w = fmaxf(v.w, 0.0f);
  }
  float ss = v.x*v.x + v.y*v.y + v.z*v.z + v.w*v.w;
  ss = red32(ss);
  float n = fmaxf(sqrtf(ss), EPSF);
  float f = artanhf_(fminf(n, MAXNF)) / n;
  v4bf o;
  o.x = (__bf16)(v.x * f); o.y = (__bf16)(v.y * f);
  o.z = (__bf16)(v.z * f); o.w = (__bf16)(v.w * f);
  t[(size_t)wave * (DIMN/4) + lane] = o;
}

// ---------------- WMMA GEMM (h = T @ W^T + b) fused with expmap0 -----------
// Wave computes a 16-row x 128-col tile. A: 16x32 bf16 fragments per the ISA
// layout; B(k,j)=W[j][k] so each lane reads two 8-half runs of row j, 16 halves apart.
__global__ __launch_bounds__(256) void gemm_expmap_wmma(
    const __bf16* __restrict__ T, const __bf16* __restrict__ Wb,
    const float* __restrict__ bias, float* __restrict__ Y, int n_nodes) {
  int wave_in_blk = threadIdx.x >> 5;
  int lane        = threadIdx.x & 31;
  int row_base    = (blockIdx.x * 8 + wave_in_blk) * 16;
  if (row_base >= n_nodes) return;           // wave-uniform: EXEC stays all-1s
  int lanelo = lane & 15;
  int hi     = lane >> 4;

  v8f acc[8];
  #pragma unroll
  for (int jt = 0; jt < 8; ++jt) acc[jt] = (v8f){0,0,0,0,0,0,0,0};

  int arow = row_base + lanelo;
  if (arow > n_nodes - 1) arow = n_nodes - 1;          // clamp; masked at store
  const __bf16* aptr = T + (size_t)arow * DIMN;

  #pragma unroll
  for (int kc = 0; kc < 4; ++kc) {
    // A fragment: lane<16 -> K = kc*32 + {0..7, 16..23}; lane>=16 -> +8
    v4u a0 = *(const v4u*)(aptr + kc * 32 +      hi * 8);
    v4u a1 = *(const v4u*)(aptr + kc * 32 + 16 + hi * 8);
    v8u au = {a0.x, a0.y, a0.z, a0.w, a1.x, a1.y, a1.z, a1.w};
    v16bf afrag = __builtin_bit_cast(v16bf, au);
    #pragma unroll
    for (int jt = 0; jt < 8; ++jt) {
      // B fragment: column j = jt*16 + lanelo; halves K = kc*32 + hi*8 + {0..7, 16..23}
      const __bf16* bptr = Wb + (size_t)(jt * 16 + lanelo) * DIMN + kc * 32 + hi * 8;
      v4u b0 = *(const v4u*)(bptr);
      v4u b1 = *(const v4u*)(bptr + 16);
      v8u bu = {b0.x, b0.y, b0.z, b0.w, b1.x, b1.y, b1.z, b1.w};
      v16bf bfrag = __builtin_bit_cast(v16bf, bu);
      acc[jt] = __builtin_amdgcn_wmma_f32_16x16x32_bf16(
          false, afrag, false, bfrag, (short)0, acc[jt], false, false);
    }
  }

  // Epilogue: +bias, row sum-of-squares (C/D layout: VGPR v <-> row v+8*hi),
  // expmap0 scale = tanh(n)/n, store y.
  float partial[8] = {0,0,0,0,0,0,0,0};
  #pragma unroll
  for (int jt = 0; jt < 8; ++jt) {
    float bj = bias[jt * 16 + lanelo];
    #pragma unroll
    for (int v = 0; v < 8; ++v) {
      float h = acc[jt][v] + bj;
      acc[jt][v] = h;
      partial[v] += h * h;
    }
  }
  float scale[8];
  #pragma unroll
  for (int v = 0; v < 8; ++v) {
    float ss = red16(partial[v]);            // row lives in one 16-lane half
    float n  = fmaxf(sqrtf(ss), EPSF);
    scale[v] = tanhf(n) / n;
  }
  #pragma unroll
  for (int v = 0; v < 8; ++v) {
    int row = row_base + v + 8 * hi;
    if (row < n_nodes) {
      #pragma unroll
      for (int jt = 0; jt < 8; ++jt)
        Y[(size_t)row * DIMN + jt * 16 + lanelo] = acc[jt][v] * scale[v];
    }
  }
}

// ---------------- edge scatter: acc[dst] += y[src] -------------------------
// one wave per edge, float4 per lane, f32 global atomics
__global__ void scatter_kernel(const float* __restrict__ Y,
                               const int* __restrict__ src,
                               const int* __restrict__ dst,
                               float* __restrict__ acc, int n_edges) {
  int wave = (blockIdx.x * blockDim.x + threadIdx.x) >> 5;
  int lane = threadIdx.x & 31;
  if (wave >= n_edges) return;
  int s = src[wave], d = dst[wave];
  float4 v = ((const float4*)(Y + (size_t)s * DIMN))[lane];
  float* base = acc + (size_t)d * DIMN + lane * 4;
  atomicAdd(base + 0, v.x);
  atomicAdd(base + 1, v.y);
  atomicAdd(base + 2, v.z);
  atomicAdd(base + 3, v.w);
}

// ---------------- final: relu -> logmap0 -> t @ Wc^T + bc ------------------
__global__ void classifier_kernel(const float* __restrict__ acc,
                                  const float* __restrict__ Wc,
                                  const float* __restrict__ bc,
                                  float* __restrict__ out,
                                  int n_nodes, int n_cls) {
  int wave = (blockIdx.x * blockDim.x + threadIdx.x) >> 5;
  int lane = threadIdx.x & 31;
  if (wave >= n_nodes) return;
  float4 v = ((const float4*)(acc + (size_t)wave * DIMN))[lane];
  v.x = fmaxf(v.x, 0.0f); v.y = fmaxf(v.y, 0.0f);
  v.z = fmaxf(v.z, 0.0f); v.w = fmaxf(v.w, 0.0f);
  float ss = red32(v.x*v.x + v.y*v.y + v.z*v.z + v.w*v.w);
  float n = fmaxf(sqrtf(ss), EPSF);
  float f = artanhf_(fminf(n, MAXNF)) / n;
  float4 t = { v.x*f, v.y*f, v.z*f, v.w*f };
  for (int c = 0; c < n_cls; ++c) {
    float4 w = ((const float4*)(Wc + (size_t)c * DIMN))[lane];
    float dot = t.x*w.x + t.y*w.y + t.z*w.z + t.w*w.w;
    dot = red32(dot);
    if (lane == 0) out[(size_t)wave * n_cls + c] = dot + bc[c];
  }
}

// ---------------------------------------------------------------------------
extern "C" void kernel_launch(void* const* d_in, const int* in_sizes, int n_in,
                              void* d_out, int out_size, void* d_ws, size_t ws_size,
                              hipStream_t stream) {
  const int*   ei = (const int*)d_in[0];
  const float* x  = (const float*)d_in[1];
  const float* W1 = (const float*)d_in[2];
  const float* b1 = (const float*)d_in[3];
  const float* W2 = (const float*)d_in[4];
  const float* b2 = (const float*)d_in[5];
  const float* Wc = (const float*)d_in[6];
  const float* bc = (const float*)d_in[7];
  float* out = (float*)d_out;

  const int n_edges = in_sizes[0] / 2;
  const int n_nodes = in_sizes[1] / DIMN;
  const int n_cls   = in_sizes[7];
  const int* src = ei;
  const int* dst = ei + n_edges;

  // workspace layout (256B aligned slices)
  char*  ws  = (char*)d_ws;
  size_t off = 0;
  auto slice = [&](size_t bytes) {
    size_t p = off; off += (bytes + 255) & ~(size_t)255; return (void*)(ws + p);
  };
  __bf16* Wb1 = (__bf16*)slice((size_t)DIMN * DIMN * sizeof(__bf16));
  __bf16* Wb2 = (__bf16*)slice((size_t)DIMN * DIMN * sizeof(__bf16));
  __bf16* Tt  = (__bf16*)slice((size_t)n_nodes * DIMN * sizeof(__bf16));
  float*  Y   = (float*) slice((size_t)n_nodes * DIMN * sizeof(float));
  float*  ACC = (float*) slice((size_t)n_nodes * DIMN * sizeof(float));
  (void)ws_size; (void)n_in; (void)out_size;

  const int TB = 256;                       // 8 waves / block
  const int node_blocks = (n_nodes + 7) / 8;
  const int gemm_blocks = (n_nodes + 127) / 128;
  const int edge_blocks = (n_edges + 7) / 8;

  cvt_bf16_kernel<<<(DIMN*DIMN + TB-1)/TB, TB, 0, stream>>>(W1, Wb1, DIMN*DIMN);
  cvt_bf16_kernel<<<(DIMN*DIMN + TB-1)/TB, TB, 0, stream>>>(W2, Wb2, DIMN*DIMN);

  // ---- layer 1 ----
  tangent_kernel<<<node_blocks, TB, 0, stream>>>(x, (v4bf*)Tt, n_nodes, 0);
  gemm_expmap_wmma<<<gemm_blocks, TB, 0, stream>>>(Tt, Wb1, b1, Y, n_nodes);
  hipMemsetAsync(ACC, 0, (size_t)n_nodes * DIMN * sizeof(float), stream);
  scatter_kernel<<<edge_blocks, TB, 0, stream>>>(Y, src, dst, ACC, n_edges);

  // ---- layer 2 (relu fused into tangent) ----
  tangent_kernel<<<node_blocks, TB, 0, stream>>>(ACC, (v4bf*)Tt, n_nodes, 1);
  gemm_expmap_wmma<<<gemm_blocks, TB, 0, stream>>>(Tt, Wb2, b2, Y, n_nodes);
  hipMemsetAsync(ACC, 0, (size_t)n_nodes * DIMN * sizeof(float), stream);
  scatter_kernel<<<edge_blocks, TB, 0, stream>>>(Y, src, dst, ACC, n_edges);

  // ---- classifier (relu + logmap0 + small GEMM fused) ----
  classifier_kernel<<<node_blocks, TB, 0, stream>>>(ACC, Wc, bc, out, n_nodes, n_cls);
}
